// DecoderLSTM_78494822301859
// MI455X (gfx1250) — matrix operand order repaired and checked by reading, not compiled, not run on hardware
//
#include <hip/hip_runtime.h>
#include <hip/hip_bf16.h>
#include <stdint.h>

// ---------------------------------------------------------------------------
// DecoderLSTM for MI455X (gfx1250).
//   B=128, T=256, V=1000 (padded 1008), H=C=E=512, 4H=2048.
// Multi-WGP persistent cooperative kernel: 16 blocks x 512 threads (256 waves),
// bf16 WMMA (v_wmma_f32_16x16x32_bf16) f32-accumulate, 2 global barriers/step.
// h double-buffered in L2-resident global bf16; argmax fused via packed u64
// global atomicMax; embedding GEMM hoisted to a per-token projection table.
// ---------------------------------------------------------------------------

typedef __attribute__((ext_vector_type(16))) __bf16 v16bf;
typedef __attribute__((ext_vector_type(8)))  float  v8f;

#define B_    128
#define T_    256
#define V_    1000
#define VP_   1008   // V padded to multiple of 16
#define H_    512
#define G4H_  2048
#define START_ID 0
#define NBLK  16
#define NTHR  512    // 16 waves per block -> 256 waves total

// ---- WMMA helpers ----------------------------------------------------------

__device__ __forceinline__ v8f wmma_bf16(v16bf a, v16bf b, v8f c) {
    return __builtin_amdgcn_wmma_f32_16x16x32_bf16(
        /*neg_a=*/false, a, /*neg_b=*/false, b,
        /*c_mod=*/(short)0, c, /*reuse_a=*/false, /*reuse_b=*/false);
}

// A-matrix 16x32 bf16 fragment from a row-major [row][K] array.
// lanes 0-15: row M=lane, K = {k0+0..7, k0+16..23};
// lanes 16-31: row M=lane-16, K = {k0+8..15, k0+24..31}.
__device__ __forceinline__ v16bf load_fragA(const __bf16* base, int row0, int ld, int k0) {
    const int lane = threadIdx.x & 31;
    const int r    = lane & 15;
    const int ko   = (lane >> 4) << 3;     // 0 or 8
    const __bf16* p = base + (size_t)(row0 + r) * ld + (k0 + ko);
    union { v16bf v; uint4 u[2]; } U;
    U.u[0] = *(const uint4*)(p);
    U.u[1] = *(const uint4*)(p + 16);
    return U.v;
}

// B-matrix 32x16 bf16 fragment; operand stored row-major [n][K].
// lanes 0-15: col N=lane, K = {k0+0..7, k0+16..23}; lanes 16-31: K = {k0+8..15, k0+24..31}.
__device__ __forceinline__ v16bf load_fragB(const __bf16* base, int n0, int ld, int k0) {
    const int lane = threadIdx.x & 31;
    const int n    = lane & 15;
    const int ko   = (lane >> 4) << 3;     // 0 or 8
    const __bf16* p = base + (size_t)(n0 + n) * ld + (k0 + ko);
    union { v16bf v; uint4 u[2]; } U;
    U.u[0] = *(const uint4*)(p);
    U.u[1] = *(const uint4*)(p + 16);
    return U.v;
}

// ---- order-preserving float pack for argmax-with-lowest-index tiebreak -----
__device__ __forceinline__ unsigned long long pack_max(float v, int col) {
    unsigned ub = __float_as_uint(v);
    ub = (ub & 0x80000000u) ? ~ub : (ub | 0x80000000u);             // monotonic map
    return ((unsigned long long)ub << 32) | (unsigned)(1023 - col); // lower col wins ties
}

// ---- device-wide barrier: monotonic counter, 1 atomic per block ------------
__device__ __forceinline__ void gbar(unsigned* bar, unsigned target) {
    __threadfence();
    __syncthreads();
    if (threadIdx.x == 0) {
        __hip_atomic_fetch_add(bar, 1u, __ATOMIC_ACQ_REL, __HIP_MEMORY_SCOPE_AGENT);
        while (__hip_atomic_load(bar, __ATOMIC_ACQUIRE, __HIP_MEMORY_SCOPE_AGENT) < target)
            __builtin_amdgcn_s_sleep(8);
    }
    __syncthreads();
    __threadfence();
}

// ---- one-time conversions: fp32 (sub)matrix -> bf16, zero row padding ------

__global__ void cvt_pad_bf16_k(__bf16* dst, const float* src, int rows, int cols,
                               int srcPitch, int srcColOff, int srcRows) {
    int i = blockIdx.x * blockDim.x + threadIdx.x;
    int n = rows * cols;
    if (i >= n) return;
    int r = i / cols, k = i - r * cols;
    float v = (r < srcRows) ? src[(size_t)r * srcPitch + srcColOff + k] : 0.0f;
    dst[i] = (__bf16)v;
}

// ---- per-launch state init (graph-capture safe, no memset) -----------------

__global__ void init_zero_k(float* cbuf, unsigned long long* gAm, unsigned* bar) {
    int i = blockIdx.x * blockDim.x + threadIdx.x;
    if (i < B_ * H_) cbuf[i] = 0.0f;
    if (i < 2 * B_)  gAm[i] = 0ull;
    if (i == 0)      *bar = 0u;
}

// ---- one-time GEMM: D[m][n] = sum_k A[m][k]*Bm[n][k] + bias0[n] + bias1[n] -
// One wave per 16x16 output tile. blockDim = 256 (8 waves).

__global__ void gemm_bf16_f32_k(const __bf16* __restrict__ A, const __bf16* __restrict__ Bm,
                                const float* __restrict__ bias0, const float* __restrict__ bias1,
                                float* __restrict__ D, int Ntiles, int totTiles, int K, int ldD) {
    int wave = (blockIdx.x * blockDim.x + threadIdx.x) >> 5;
    if (wave >= totTiles) return;                 // wave-uniform
    int mt = wave / Ntiles, nt = wave - mt * Ntiles;
    int lane = threadIdx.x & 31;
    int n = nt * 16 + (lane & 15);
    float binit = 0.0f;
    if (bias0) binit += bias0[n];
    if (bias1) binit += bias1[n];
    v8f acc;
#pragma unroll
    for (int r = 0; r < 8; ++r) acc[r] = binit;
    for (int k0 = 0; k0 < K; k0 += 32) {
        v16bf a = load_fragA(A,  mt * 16, K, k0);
        v16bf b = load_fragB(Bm, nt * 16, K, k0);
        acc = wmma_bf16(a, b, acc);
    }
    int mbase = mt * 16 + ((lane >> 4) << 3);
#pragma unroll
    for (int r = 0; r < 8; ++r) D[(size_t)(mbase + r) * ldD + n] = acc[r];
}

// ---- persistent cooperative sequential LSTM (16 blocks x 16 waves) ---------

__launch_bounds__(NTHR, 1)
__global__ void lstm_seq_k(const int* __restrict__ tgt, const float* __restrict__ coin,
                           const float* __restrict__ ctxproj,   // [128][2048] incl. biases
                           const float* __restrict__ embproj,   // [1008][2048]
                           const __bf16* __restrict__ Whh,      // [2048][512] bf16
                           const __bf16* __restrict__ Wout,     // [1008][512] bf16 (rows>=1000 zero)
                           const float* __restrict__ b_out,
                           float* __restrict__ cbuf,            // [128][512] f32 cell state
                           __bf16* __restrict__ h0,             // [128][512] bf16 (= bf16 context)
                           __bf16* __restrict__ h1,             // [128][512] bf16
                           unsigned long long* __restrict__ gAm,// [2][128] packed argmax
                           unsigned* __restrict__ bar,          // global barrier counter
                           float* __restrict__ outs,            // [128][256][1000]
                           float* __restrict__ out_h, float* __restrict__ out_c) {
    const int tid  = threadIdx.x;
    const int lane = tid & 31;
    const int gid  = blockIdx.x * (NTHR / 32) + (tid >> 5);   // global wave id: 0..255
    const int hi8  = (lane >> 4) << 3;

    // gates tile-group owned by this wave for the whole sequence
    const int bt    = gid >> 5;                 // batch tile 0..7
    const int nt    = gid & 31;                 // hidden column tile 0..31
    const int n     = nt * 16 + (lane & 15);    // gate-relative column 0..511
    const int mbase = bt * 16 + hi8;

    unsigned nbar = 0;

    for (int t = 0; t < T_; ++t) {
        const __bf16* hin  = (t & 1) ? h1 : h0;
        __bf16*       hout = (t & 1) ? h0 : h1;
        const unsigned long long* amPrev = gAm + (((t + 1) & 1) * B_); // written at step t-1
        unsigned long long*       amCur  = gAm + ((t & 1) * B_);

        // reset this step's argmax buffer (last read finished before step t-1's
        // end barrier; visible to all before barrier (a) below)
        if (blockIdx.x == 0 && tid < B_) amCur[tid] = 0ull;

        // ---- phase A: gates = ctxproj + embproj[tok] + h @ Whh^T ; LSTM cell
        const float cf = coin[t];              // step-uniform scheduled sampling
        v8f a0, a1, a2, a3;
#pragma unroll
        for (int r = 0; r < 8; ++r) {
            const int m = mbase + r;
            int tok;
            if (cf > 0.5f) tok = (t == 0) ? START_ID : tgt[m * T_ + (t - 1)];
            else           tok = (t == 0) ? START_ID
                                          : (int)(1023u - (unsigned)(amPrev[m] & 0xFFFFFFFFull));
            const float* cp = ctxproj + (size_t)m * G4H_;
            const float* ep = embproj + (size_t)tok * G4H_;
            a0[r] = cp[n]        + ep[n];
            a1[r] = cp[n + 512]  + ep[n + 512];
            a2[r] = cp[n + 1024] + ep[n + 1024];
            a3[r] = cp[n + 1536] + ep[n + 1536];
        }
#pragma unroll 4
        for (int k0 = 0; k0 < H_; k0 += 32) {
            v16bf a  = load_fragA(hin, bt * 16, H_, k0);
            v16bf b0 = load_fragB(Whh, 0 * 512 + nt * 16, H_, k0);
            v16bf b1 = load_fragB(Whh, 1 * 512 + nt * 16, H_, k0);
            v16bf b2 = load_fragB(Whh, 2 * 512 + nt * 16, H_, k0);
            v16bf b3 = load_fragB(Whh, 3 * 512 + nt * 16, H_, k0);
            a0 = wmma_bf16(a, b0, a0);
            a1 = wmma_bf16(a, b1, a1);
            a2 = wmma_bf16(a, b2, a2);
            a3 = wmma_bf16(a, b3, a3);
        }
#pragma unroll
        for (int r = 0; r < 8; ++r) {
            const int m = mbase + r;
            float ig = 1.0f / (1.0f + __expf(-a0[r]));
            float fg = 1.0f / (1.0f + __expf(-a1[r]));
            float gg = tanhf(a2[r]);
            float og = 1.0f / (1.0f + __expf(-a3[r]));
            const size_t ci = (size_t)m * H_ + n;
            float cn = fg * cbuf[ci] + ig * gg;   // this thread wrote cbuf[ci] last step
            float hn = og * tanhf(cn);
            cbuf[ci] = cn;
            hout[ci] = (__bf16)hn;
            if (t == T_ - 1) { out_h[ci] = hn; out_c[ci] = cn; }
        }

        gbar(bar, NBLK * ++nbar);   // (a) h_new complete device-wide

        // ---- phase B: logits = h_new @ Wout^T + b_out ; fused argmax
        // 504 tiles = 8 batch-tiles x 63 vocab-tiles, strided over 256 waves
#pragma unroll 1
        for (int id = gid; id < 504; id += 256) {
            const int bt2 = id & 7;
            const int vt  = id >> 3;
            const int col = vt * 16 + (lane & 15);
            v8f acc;
            const float binit = (col < V_) ? b_out[col] : -1e30f;
#pragma unroll
            for (int r = 0; r < 8; ++r) acc[r] = binit;
#pragma unroll 4
            for (int k0 = 0; k0 < H_; k0 += 32) {
                v16bf a = load_fragA(hout, bt2 * 16, H_, k0);
                v16bf b = load_fragB(Wout, vt * 16, H_, k0);
                acc = wmma_bf16(a, b, acc);
            }
            const int mb2 = bt2 * 16 + hi8;
#pragma unroll
            for (int r = 0; r < 8; ++r) {
                const int m = mb2 + r;
                if (col < V_) outs[((size_t)m * T_ + t) * V_ + col] = acc[r];
                // 16-lane (half-wave) argmax for row m over this 16-col slice
                float v = acc[r]; int c = col;
#pragma unroll
                for (int s = 1; s < 16; s <<= 1) {
                    float ov = __shfl_xor(v, s, 16);
                    int   oc = __shfl_xor(c, s, 16);
                    if (ov > v || (ov == v && oc < c)) { v = ov; c = oc; }
                }
                if ((lane & 15) == 0)
                    atomicMax(&amCur[m], pack_max(v, c));   // global_atomic_max_u64
            }
        }

        gbar(bar, NBLK * ++nbar);   // (b) argmax complete; end of step
    }
}

// ---------------------------------------------------------------------------

extern "C" void kernel_launch(void* const* d_in, const int* in_sizes, int n_in,
                              void* d_out, int out_size, void* d_ws, size_t ws_size,
                              hipStream_t stream) {
    const float* context = (const float*)d_in[0];
    const int*   target  = (const int*)  d_in[1];
    const float* coin    = (const float*)d_in[2];
    const float* embed   = (const float*)d_in[3];
    const float* W_ih    = (const float*)d_in[4];   // [2048][1024]
    const float* W_hh    = (const float*)d_in[5];   // [2048][512]
    const float* b_ih    = (const float*)d_in[6];
    const float* b_hh    = (const float*)d_in[7];
    const float* W_out   = (const float*)d_in[8];   // [1000][512]
    const float* b_out   = (const float*)d_in[9];

    // workspace carve (256B aligned)
    char* ws = (char*)d_ws; size_t off = 0;
    auto carve = [&](size_t bytes) -> void* {
        void* p = ws + off; off += (bytes + 255) & ~(size_t)255; return p;
    };
    unsigned*           bar  = (unsigned*)           carve(256);
    unsigned long long* gAm  = (unsigned long long*) carve(2 * B_ * 8);
    float*  cbuf    = (float*) carve((size_t)B_ * H_ * 4);
    float*  ctxproj = (float*) carve((size_t)B_ * G4H_ * 4);
    float*  embproj = (float*) carve((size_t)VP_ * G4H_ * 4);
    __bf16* whhB    = (__bf16*)carve((size_t)G4H_ * H_ * 2);
    __bf16* woutB   = (__bf16*)carve((size_t)VP_ * H_ * 2);
    __bf16* h0buf   = (__bf16*)carve((size_t)B_ * H_ * 2);   // bf16 context == h at t=0
    __bf16* h1buf   = (__bf16*)carve((size_t)B_ * H_ * 2);
    __bf16* wihC    = (__bf16*)carve((size_t)G4H_ * H_ * 2);
    __bf16* wihE    = (__bf16*)carve((size_t)G4H_ * H_ * 2);
    __bf16* embB    = (__bf16*)carve((size_t)VP_ * H_ * 2);

    hipLaunchKernelGGL(init_zero_k, dim3((B_ * H_ + 255) / 256), dim3(256), 0, stream,
                       cbuf, gAm, bar);

    auto cvt = [&](__bf16* dst, const float* src, int rows, int cols,
                   int pitch, int coff, int srows) {
        int n = rows * cols;
        hipLaunchKernelGGL(cvt_pad_bf16_k, dim3((n + 255) / 256), dim3(256), 0, stream,
                           dst, src, rows, cols, pitch, coff, srows);
    };
    cvt(whhB,  W_hh,    G4H_, H_, H_,   0,   G4H_);  // W_hh -> bf16
    cvt(woutB, W_out,   VP_,  H_, H_,   0,   V_);    // W_out -> bf16, zero-pad rows
    cvt(h0buf, context, B_,   H_, H_,   0,   B_);    // context -> bf16 (= h0, gemm A)
    cvt(wihC,  W_ih,    G4H_, H_, 1024, 0,   G4H_);  // W_ih[:, :512]
    cvt(wihE,  W_ih,    G4H_, H_, 1024, 512, G4H_);  // W_ih[:, 512:]
    cvt(embB,  embed,   VP_,  H_, H_,   0,   V_);    // embed_table -> bf16, pad

    // ctxproj[128][2048] = context @ W_ihC^T + b_ih + b_hh   (8*128 = 1024 tiles)
    hipLaunchKernelGGL(gemm_bf16_f32_k, dim3(128), dim3(256), 0, stream,
                       h0buf, wihC, b_ih, b_hh, ctxproj, 128, 1024, H_, G4H_);
    // embproj[1008][2048] = embed @ W_ihE^T                  (63*128 = 8064 tiles)
    hipLaunchKernelGGL(gemm_bf16_f32_k, dim3(1008), dim3(256), 0, stream,
                       embB, wihE, (const float*)nullptr, (const float*)nullptr,
                       embproj, 128, 8064, H_, G4H_);

    float* outs  = (float*)d_out;
    float* out_h = outs + (size_t)B_ * T_ * V_;
    float* out_c = out_h + (size_t)B_ * H_;
    hipLaunchKernelGGL(lstm_seq_k, dim3(NBLK), dim3(NTHR), 0, stream,
                       target, coin, ctxproj, embproj, whhB, woutB, b_out,
                       cbuf, h0buf, h1buf, gAm, bar, outs, out_h, out_c);
}
